// TestSiluMulFp8QuantModel_24352464569816
// MI455X (gfx1250) — 8-bit matrix operand order repaired and checked by reading, compiled, not run on hardware
//
#include <hip/hip_runtime.h>

#define T_DIM 8192
#define H_DIM 4096
#define FP8_MAX 448.0f

#define LDS_STRIDE 144

typedef __attribute__((ext_vector_type(16))) int   v16i;
typedef __attribute__((ext_vector_type(8)))  float v8f;

__device__ inline float clamp448(float v) {
    return fminf(fmaxf(v, -FP8_MAX), FP8_MAX);
}

__device__ inline unsigned int f32_to_e4m3_sw(float f) {
    unsigned int u = __float_as_uint(f);
    unsigned int sign = (u >> 24) & 0x80u;
    float af = fabsf(f);
    if (af > FP8_MAX) af = FP8_MAX;
    if (af < 0.0009765625f) return sign;
    u = __float_as_uint(af);
    int exp = (int)(u >> 23) - 127;
    unsigned int mant = u & 0x7FFFFFu;
    unsigned int out;
    if (exp < -6) {
        int shift = -6 - exp;
        mant |= 0x800000u;
        unsigned int m    = mant >> (20 + shift);
        unsigned int rem  = mant & ((1u << (20 + shift)) - 1u);
        unsigned int half = 1u << (19 + shift);
        if (rem > half || (rem == half && (m & 1u))) m++;
        out = m;
    } else {
        unsigned int m   = mant >> 20;
        unsigned int rem = mant & 0xFFFFFu;
        if (rem > 0x80000u || (rem == 0x80000u && (m & 1u))) m++;
        unsigned int v = (((unsigned int)(exp + 7)) << 3) + m;
        if (v > 0x7Eu) v = 0x7Eu;
        out = v;
    }
    return sign | out;
}

__device__ inline unsigned int pack4_e4m3(float a, float b, float c, float d) {
#if __has_builtin(__builtin_amdgcn_cvt_pk_fp8_f32)
    int p = 0;
    p = __builtin_amdgcn_cvt_pk_fp8_f32(a, b, p, false);
    p = __builtin_amdgcn_cvt_pk_fp8_f32(c, d, p, true);
    return (unsigned int)p;
#else
    return  f32_to_e4m3_sw(a)
         | (f32_to_e4m3_sw(b) << 8)
         | (f32_to_e4m3_sw(c) << 16)
         | (f32_to_e4m3_sw(d) << 24);
#endif
}

__device__ inline unsigned int lds_off(const void* p) {
    return (unsigned int)(size_t)p;
}

__device__ inline void async_load_b128(unsigned int lds_addr, const void* gaddr) {
    asm volatile("global_load_async_to_lds_b128 %0, %1, off"
                 :: "v"(lds_addr), "v"(gaddr) : "memory");
}

__device__ inline void wait_async0() {
    asm volatile("s_wait_asynccnt 0" ::: "memory");
}

__global__ __launch_bounds__(256) void silu_mul_quant_kernel(
    const float* __restrict__ x, const float* __restrict__ wscale,
    unsigned char* __restrict__ q)
{
    size_t i4 = ((size_t)blockIdx.x * 256 + threadIdx.x) * 4;
    size_t t  = i4 >> 12;
    size_t h  = i4 & 4095;
    const float* xr = x + t * (size_t)(2 * H_DIM);

    float4 g = *(const float4*)(xr + h);
    float4 u = *(const float4*)(xr + H_DIM + h);

    float inv_s = 1.0f / wscale[0];

    float y0 = (g.x / (1.0f + __expf(-g.x))) * u.x * inv_s;
    float y1 = (g.y / (1.0f + __expf(-g.y))) * u.y * inv_s;
    float y2 = (g.z / (1.0f + __expf(-g.z))) * u.z * inv_s;
    float y3 = (g.w / (1.0f + __expf(-g.w))) * u.w * inv_s;

    *(unsigned int*)(q + i4) =
        pack4_e4m3(clamp448(y0), clamp448(y1), clamp448(y2), clamp448(y3));
}

__global__ __launch_bounds__(256) void wquant_kernel(
    const float* __restrict__ w, unsigned char* __restrict__ wqt)
{
    int k = blockIdx.y;
    int n = (blockIdx.x * 256 + threadIdx.x) * 4;

    float4 v = *(const float4*)(w + (size_t)k * H_DIM + n);
    unsigned int p = pack4_e4m3(clamp448(v.x), clamp448(v.y),
                                clamp448(v.z), clamp448(v.w));
    wqt[(size_t)(n + 0) * H_DIM + k] = (unsigned char)(p       & 0xFF);
    wqt[(size_t)(n + 1) * H_DIM + k] = (unsigned char)((p>>8)  & 0xFF);
    wqt[(size_t)(n + 2) * H_DIM + k] = (unsigned char)((p>>16) & 0xFF);
    wqt[(size_t)(n + 3) * H_DIM + k] = (unsigned char)((p>>24) & 0xFF);
}

__global__ __launch_bounds__(256) void gemm_fp8_kernel(
    const unsigned char* __restrict__ A,
    const unsigned char* __restrict__ Bt,
    const float* __restrict__ wscale,
    float* __restrict__ out)
{
    __shared__ __align__(16) unsigned char lsA[2][128 * LDS_STRIDE];
    __shared__ __align__(16) unsigned char lsB[2][128 * LDS_STRIDE];

    const int tid   = threadIdx.x;
    const int lane  = tid & 31;
    const int wave  = tid >> 5;
    const int laneL = lane & 15;
    const int hi    = lane >> 4;

    const int gm0 = blockIdx.y * 128;
    const int gn0 = blockIdx.x * 128;
    const int wm  = wave & 3;
    const int wn  = wave >> 2;

    const v8f vzero = {0.f, 0.f, 0.f, 0.f, 0.f, 0.f, 0.f, 0.f};
    v8f acc[2][4];
#pragma unroll
    for (int i = 0; i < 2; ++i)
#pragma unroll
        for (int j = 0; j < 4; ++j) acc[i][j] = vzero;

    const int ldRow0 = tid >> 3;
    const int ldCol  = (tid & 7) * 16;

    const unsigned char* gA = A  + (size_t)gm0 * H_DIM;
    const unsigned char* gB = Bt + (size_t)gn0 * H_DIM;

    unsigned int lA[2] = { lds_off(&lsA[0][0]), lds_off(&lsA[1][0]) };
    unsigned int lB[2] = { lds_off(&lsB[0][0]), lds_off(&lsB[1][0]) };

#pragma unroll
    for (int it = 0; it < 4; ++it) {
        int r = it * 32 + ldRow0;
        async_load_b128(lA[0] + r * LDS_STRIDE + ldCol,
                        gA + (size_t)r * H_DIM + ldCol);
        async_load_b128(lB[0] + r * LDS_STRIDE + ldCol,
                        gB + (size_t)r * H_DIM + ldCol);
    }
    wait_async0();
    __syncthreads();

    for (int kt = 0; kt < H_DIM / 128; ++kt) {
        const int cur = kt & 1;

        if (kt + 1 < H_DIM / 128) {
            const int k0n = (kt + 1) * 128;
#pragma unroll
            for (int it = 0; it < 4; ++it) {
                int r = it * 32 + ldRow0;
                async_load_b128(lA[cur ^ 1] + r * LDS_STRIDE + ldCol,
                                gA + (size_t)r * H_DIM + k0n + ldCol);
                async_load_b128(lB[cur ^ 1] + r * LDS_STRIDE + ldCol,
                                gB + (size_t)r * H_DIM + k0n + ldCol);
            }
        }

        v16i afrag[2];
#pragma unroll
        for (int i = 0; i < 2; ++i) {
            const unsigned char* base =
                &lsA[cur][(wm * 32 + i * 16 + laneL) * LDS_STRIDE];
            union { v16i v; unsigned long long q[8]; } ua;
#pragma unroll
            for (int half = 0; half < 2; ++half)
#pragma unroll
                for (int j = 0; j < 4; ++j)
                    ua.q[half * 4 + j] = *(const unsigned long long*)
                        (base + half * 64 + (j >> 1) * 32 + hi * 16 + (j & 1) * 8);
            afrag[i] = ua.v;
        }

        v16i bfrag[4];
#pragma unroll
        for (int j = 0; j < 4; ++j) {
            const unsigned char* base =
                &lsB[cur][(wn * 64 + j * 16 + laneL) * LDS_STRIDE];
            union { v16i v; int4 x[4]; } ub;
#pragma unroll
            for (int g = 0; g < 4; ++g)
                ub.x[g] = *(const int4*)(base + g * 32 + hi * 16);
            bfrag[j] = ub.v;
        }

#pragma unroll
        for (int j = 0; j < 4; ++j) {
            acc[0][j] = __builtin_amdgcn_wmma_f32_16x16x128_fp8_fp8(
                afrag[0], bfrag[j], (short)0, acc[0][j], false, false);
            acc[1][j] = __builtin_amdgcn_wmma_f32_16x16x128_fp8_fp8(
                afrag[1], bfrag[j], (short)0, acc[1][j], false, false);
        }

        wait_async0();
        __syncthreads();
    }

    float s  = wscale[0];
    float s2 = s * s;
#pragma unroll
    for (int i = 0; i < 2; ++i) {
        int rowBase = gm0 + wm * 32 + i * 16 + hi * 8;
#pragma unroll
        for (int j = 0; j < 4; ++j) {
            int col = gn0 + wn * 64 + j * 16 + laneL;
#pragma unroll
            for (int r = 0; r < 8; ++r)
                out[(size_t)(rowBase + r) * H_DIM + col] = acc[i][j][r] * s2;
        }
    }
}

extern "C" void kernel_launch(void* const* d_in, const int* in_sizes, int n_in,
                              void* d_out, int out_size, void* d_ws, size_t ws_size,
                              hipStream_t stream)
{
    const float* x      = (const float*)d_in[0];
    const float* w      = (const float*)d_in[1];
    const float* wscale = (const float*)d_in[2];
    float*       out    = (float*)d_out;

    unsigned char* q   = (unsigned char*)d_ws;
    unsigned char* wqt = q + (size_t)T_DIM * H_DIM;

    silu_mul_quant_kernel<<<(T_DIM * (size_t)H_DIM) / (256 * 4), 256, 0, stream>>>(
        x, wscale, q);

    wquant_kernel<<<dim3(H_DIM / (256 * 4), H_DIM), 256, 0, stream>>>(w, wqt);

    gemm_fp8_kernel<<<dim3(H_DIM / 128, T_DIM / 128), 256, 0, stream>>>(
        q, wqt, wscale, out);
}
